// Coarse2FineAutoRegressor_29566554866135
// MI455X (gfx1250) — compile-verified, matrix operand order checked by reading, not run on hardware
//
#include <hip/hip_runtime.h>

// ---------------------------------------------------------------------------
// Types for CDNA5 WMMA (wave32, 16x16x32 bf16 -> f32)
// ---------------------------------------------------------------------------
typedef __bf16 bf16;
typedef bf16  v16bf __attribute__((ext_vector_type(16)));
typedef bf16  v8bf  __attribute__((ext_vector_type(8)));
typedef float v8f   __attribute__((ext_vector_type(8)));
typedef float v4f   __attribute__((ext_vector_type(4)));

__device__ __forceinline__ v8f wmma_bf16(v16bf a, v16bf b, v8f c) {
  return __builtin_amdgcn_wmma_f32_16x16x32_bf16(false, a, false, b,
                                                 (short)0, c, false, false);
}

// Per ISA §10.2 the low 32 bits of a flat LDS address are the wave-relative
// LDS byte offset -> usable directly as the VDST operand of async-to-LDS ops.
__device__ __forceinline__ unsigned lds_off(const void* p) {
  return (unsigned)(size_t)p;
}

// Async copy 32B global -> LDS (2 x b128), tracked by ASYNCcnt.
__device__ __forceinline__ void async_copy32(unsigned lds, const void* gptr) {
  asm volatile(
      "global_load_async_to_lds_b128 %0, %1, off\n\t"
      "global_load_async_to_lds_b128 %0, %1, off offset:16"
      :
      : "v"(lds), "v"(gptr)
      : "memory");
}
__device__ __forceinline__ void wait_async0() {
  asm volatile("s_wait_asynccnt 0" ::: "memory");
}

// Build a 16x32 A/B fragment for this lane from a row of bf16 data.
// Lanes 0-15 hold K {0..7,16..23}, lanes 16-31 hold K {8..15,24..31}.
__device__ __forceinline__ v16bf make_frag(const bf16* __restrict__ rowbase,
                                           int lanehi) {
  const v8bf* p0 = (const v8bf*)(rowbase + (lanehi ? 8 : 0));
  const v8bf* p1 = (const v8bf*)(rowbase + 16 + (lanehi ? 8 : 0));
  v8bf a = *p0, b = *p1;
  v16bf f;
#pragma unroll
  for (int e = 0; e < 8; ++e) { f[e] = a[e]; f[e + 8] = b[e]; }
  return f;
}

__device__ __forceinline__ unsigned pack_bf16(float lo, float hi) {
  unsigned l = __builtin_bit_cast(unsigned short, (bf16)lo);
  unsigned h = __builtin_bit_cast(unsigned short, (bf16)hi);
  return l | (h << 16);
}

// ---------------------------------------------------------------------------
// Generic WMMA GEMM:  out = EPI( A(bf16, MxK) @ W(f32, KxN) + bias )
// Block tile 128x128, K-step 32, 8 waves of 64x32 each.
// MODE 0: store bf16      MODE 1: gelu(tanh) -> bf16
// MODE 2: resid[r,c] += (acc+bias)*gmul[(r/100)*g_stride+g_ofs+c]
// MODE 3: store f32
// ---------------------------------------------------------------------------
#define BM 128
#define BN 128
#define BK 32

template <int MODE>
__global__ __launch_bounds__(256) void gemm_wmma(
    const bf16* __restrict__ A, const float* __restrict__ W,
    const float* __restrict__ bias, void* __restrict__ out,
    float* __restrict__ resid, const float* __restrict__ gmul,
    int M, int N, int K, int ldo, int g_stride, int g_ofs) {
  __shared__ bf16 aT[BM * BK];   // [m][k]
  __shared__ bf16 bT[BN * BK];   // [n][k]  (transposed for contiguous frags)

  const int t = threadIdx.x;
  const int lane = t & 31, wid = t >> 5;
  const int lanehi = lane >> 4, lmod = lane & 15;
  const int waveM = wid & 1, waveN = wid >> 1;
  const int bm = blockIdx.y * BM, bn = blockIdx.x * BN;

  v8f acc[4][2];
#pragma unroll
  for (int mi = 0; mi < 4; ++mi)
#pragma unroll
    for (int nj = 0; nj < 2; ++nj) acc[mi][nj] = {};

  const int arow = t >> 1, ac0 = (t & 1) * 16;   // A: 128 rows x 32 cols
  const int kp = t & 15, n0 = (t >> 4) * 8;      // B: k-pair x 8 cols / thread
  const unsigned aLds = lds_off(&aT[arow * BK + ac0]);

  for (int k0 = 0; k0 < K; k0 += BK) {
    // --- A tile: async DMA global -> LDS (bf16 already in final layout) ---
    async_copy32(aLds, A + (size_t)(bm + arow) * K + k0 + ac0);

    // --- B tile: f32 global -> packed bf16 pairs in LDS [n][k] ---
    {
      const float* s0 = W + (size_t)(k0 + 2 * kp) * N + bn + n0;
      const float* s1 = s0 + N;
      v4f a0 = *(const v4f*)s0, a1 = *(const v4f*)(s0 + 4);
      v4f b0 = *(const v4f*)s1, b1 = *(const v4f*)(s1 + 4);
#pragma unroll
      for (int e = 0; e < 4; ++e) {
        *(unsigned*)&bT[(n0 + e) * BK + 2 * kp]     = pack_bf16(a0[e], b0[e]);
        *(unsigned*)&bT[(n0 + e + 4) * BK + 2 * kp] = pack_bf16(a1[e], b1[e]);
      }
    }
    if (k0 + BK < K) {   // hint the next K-tile into cache (global_prefetch)
      __builtin_prefetch(A + (size_t)(bm + arow) * K + k0 + BK + ac0, 0, 0);
      __builtin_prefetch(W + (size_t)(k0 + BK + 2 * kp) * N + bn + n0, 0, 0);
    }
    wait_async0();
    __syncthreads();

    v16bf bfrag[2];
#pragma unroll
    for (int nj = 0; nj < 2; ++nj)
      bfrag[nj] = make_frag(&bT[(waveN * 32 + nj * 16 + lmod) * BK], lanehi);
#pragma unroll
    for (int mi = 0; mi < 4; ++mi) {
      v16bf afrag = make_frag(&aT[(waveM * 64 + mi * 16 + lmod) * BK], lanehi);
#pragma unroll
      for (int nj = 0; nj < 2; ++nj)
        acc[mi][nj] = wmma_bf16(afrag, bfrag[nj], acc[mi][nj]);
    }
    __syncthreads();
  }

  // Epilogue. C/D layout: lane holds N = lmod, M = v + 8*lanehi per subtile.
#pragma unroll
  for (int mi = 0; mi < 4; ++mi)
#pragma unroll
    for (int nj = 0; nj < 2; ++nj)
#pragma unroll
      for (int v = 0; v < 8; ++v) {
        int row = bm + waveM * 64 + mi * 16 + lanehi * 8 + v;
        int col = bn + waveN * 32 + nj * 16 + lmod;
        float val = acc[mi][nj][v] + bias[col];
        if (MODE == 0) {
          ((bf16*)out)[(size_t)row * ldo + col] = (bf16)val;
        } else if (MODE == 1) {
          float u = 0.7978845608028654f * (val + 0.044715f * val * val * val);
          float th = 1.0f - 2.0f / (__expf(2.0f * u) + 1.0f);  // tanh(u)
          ((bf16*)out)[(size_t)row * ldo + col] = (bf16)(0.5f * val * (1.0f + th));
        } else if (MODE == 2) {
          float g = gmul[(size_t)(row / 100) * g_stride + g_ofs + col];
          resid[(size_t)row * ldo + col] += val * g;
        } else {
          ((float*)out)[(size_t)row * ldo + col] = val;
        }
      }
}

// ---------------------------------------------------------------------------
// LayerNorm + AdaLN modulate:  h = ln(x) * (scale+1) + shift   (bf16 out)
// ---------------------------------------------------------------------------
__global__ __launch_bounds__(256) void ln_mod(
    const float* __restrict__ x, bf16* __restrict__ h,
    const float* __restrict__ smat, int rowdiv, int stride, int sofs,
    int shofs) {
  const int row = blockIdx.x, t = threadIdx.x;
  const float* xr = x + (size_t)row * 768;
  __shared__ float red[256];
  float v0 = xr[t], v1 = xr[t + 256], v2 = xr[t + 512];
  red[t] = v0 + v1 + v2;
  __syncthreads();
  for (int o = 128; o > 0; o >>= 1) {
    if (t < o) red[t] += red[t + o];
    __syncthreads();
  }
  float mean = red[0] * (1.0f / 768.0f);
  __syncthreads();
  float d0 = v0 - mean, d1 = v1 - mean, d2 = v2 - mean;
  red[t] = d0 * d0 + d1 * d1 + d2 * d2;
  __syncthreads();
  for (int o = 128; o > 0; o >>= 1) {
    if (t < o) red[t] += red[t + o];
    __syncthreads();
  }
  float rstd = rsqrtf(red[0] * (1.0f / 768.0f) + 1e-6f);
  const float* sb = smat + (size_t)(row / rowdiv) * stride;
  bf16* hr = h + (size_t)row * 768;
  hr[t]       = (bf16)(d0 * rstd * (sb[sofs + t] + 1.f)       + sb[shofs + t]);
  hr[t + 256] = (bf16)(d1 * rstd * (sb[sofs + t + 256] + 1.f) + sb[shofs + t + 256]);
  hr[t + 512] = (bf16)(d2 * rstd * (sb[sofs + t + 512] + 1.f) + sb[shofs + t + 512]);
}

// ---------------------------------------------------------------------------
// cond = nobs @ obs_w + obs_b ;  cond_s = silu(cond) (bf16)
// ---------------------------------------------------------------------------
__global__ __launch_bounds__(256) void cond_kernel(
    const float* __restrict__ nobs, const float* __restrict__ obs_w,
    const float* __restrict__ obs_b, float* __restrict__ cond,
    bf16* __restrict__ conds) {
  const int b = blockIdx.x, t = threadIdx.x;
  __shared__ float f[23];
  if (t < 23) f[t] = nobs[b * 23 + t];
  __syncthreads();
  for (int c = t; c < 768; c += 256) {
    float s = obs_b[c];
#pragma unroll
    for (int k = 0; k < 23; ++k) s += f[k] * obs_w[k * 768 + c];
    cond[b * 768 + c] = s;
    conds[b * 768 + c] = (bf16)(s / (1.0f + __expf(-s)));
  }
}

__device__ __forceinline__ int lvl_of(int l) {
  return l < 10 ? 0 : (l < 30 ? 1 : (l < 60 ? 2 : 3));
}

// ---------------------------------------------------------------------------
// x[b,l,:] = (l<10 ? cond+pos_start : xwo@word_w+word_b) + lvl_emb + pos_1LC
// ---------------------------------------------------------------------------
__global__ __launch_bounds__(256) void embed_kernel(
    const float* __restrict__ xwo, const float* __restrict__ word_w,
    const float* __restrict__ word_b, const float* __restrict__ cond,
    const float* __restrict__ pos_start, const float* __restrict__ pos_1LC,
    const float* __restrict__ lvl_emb, float* __restrict__ x) {
  const int row = blockIdx.x;            // b*100 + l
  const int b = row / 100, l = row % 100, t = threadIdx.x;
  const int lv = lvl_of(l);
  __shared__ float w[32];
  if (l >= 10 && t < 32) w[t] = xwo[((size_t)b * 90 + (l - 10)) * 32 + t];
  __syncthreads();
  for (int c = t; c < 768; c += 256) {
    float v;
    if (l < 10) {
      v = cond[b * 768 + c] + pos_start[l * 768 + c];
    } else {
      v = word_b[c];
#pragma unroll
      for (int k = 0; k < 32; ++k) v += w[k] * word_w[k * 768 + c];
    }
    v += lvl_emb[lv * 768 + c] + pos_1LC[l * 768 + c];
    x[(size_t)row * 768 + c] = v;
  }
}

// ---------------------------------------------------------------------------
// Attention: one block per (batch, head). 7 waves; wave w owns queries
// [16w,16w+16). QK^T and P@V both via bf16 WMMA; softmax with lane shuffles.
// qkv layout: (B,L,3,NH,HD) -> token row stride 2304.  o: (B,L,C) bf16.
// ---------------------------------------------------------------------------
__global__ __launch_bounds__(224) void attn_kernel(
    const bf16* __restrict__ qkv, bf16* __restrict__ o) {
  __shared__ bf16 Kt[112 * 64];         // [key][hd]
  __shared__ bf16 Vt[64 * 128];         // [hd][key]  (keys padded to 128)
  __shared__ bf16 P[7 * 16 * 128];      // per-wave probs [q][key]

  const int bh = blockIdx.x;
  const int b = bh / 12, h = bh % 12;
  const int t = threadIdx.x, lane = t & 31, wid = t >> 5;
  const int lanehi = lane >> 4, lmod = lane & 15;
  const bf16* base = qkv + (size_t)b * 100 * 2304;

  // ---- stage K (async DMA, row-major) and V (transposed), zero-pad ----
  {
    const int key = t >> 1, d0 = (t & 1) * 32;
    if (key < 100) {
      async_copy32(lds_off(&Kt[key * 64 + d0]),
                   base + (size_t)key * 2304 + 768 + h * 64 + d0);
      const bf16* vs = base + (size_t)key * 2304 + 1536 + h * 64 + d0;
#pragma unroll
      for (int j = 0; j < 32; ++j) Vt[(d0 + j) * 128 + key] = vs[j];
    } else {
      v8bf z = {};
#pragma unroll
      for (int j = 0; j < 32; j += 8) *(v8bf*)&Kt[key * 64 + d0 + j] = z;
#pragma unroll
      for (int j = 0; j < 32; ++j) Vt[(d0 + j) * 128 + key] = (bf16)0.f;
    }
    for (int idx = t; idx < 64 * 16; idx += 224)   // keys 112..127
      Vt[(idx >> 4) * 128 + 112 + (idx & 15)] = (bf16)0.f;
  }
  wait_async0();
  __syncthreads();

  // ---- Q fragments (direct from global, clamped rows for the pad tile) ----
  const int q0 = wid * 16;
  int qrow = q0 + lmod; if (qrow > 99) qrow = 99;
  const bf16* qp = base + (size_t)qrow * 2304 + h * 64;
  v16bf qf0 = make_frag(qp, lanehi);
  v16bf qf1 = make_frag(qp + 32, lanehi);

  // ---- S = Q @ K^T ----
  v8f s[7];
#pragma unroll
  for (int kt = 0; kt < 7; ++kt) {
    const bf16* kb = &Kt[(kt * 16 + lmod) * 64];
    v16bf kf0 = make_frag(kb, lanehi);
    v16bf kf1 = make_frag(kb + 32, lanehi);
    v8f a = {};
    a = wmma_bf16(qf0, kf0, a);
    a = wmma_bf16(qf1, kf1, a);
    s[kt] = a;
  }

  // ---- masked softmax (rows spread over 16-lane groups + 7 regs/lane) ----
  const float scale = 0.125f;            // HD^-0.5, HD=64
#pragma unroll
  for (int v = 0; v < 8; ++v) {
    const int q = q0 + lanehi * 8 + v;
    const int lq = lvl_of(q);
    float m = -1e30f;
#pragma unroll
    for (int kt = 0; kt < 7; ++kt) {
      const int k = kt * 16 + lmod;
      bool ok = (k < 100) && (lq >= lvl_of(k));
      float val = ok ? s[kt][v] * scale : -1e30f;
      s[kt][v] = val;
      m = fmaxf(m, val);
    }
    for (int ofs = 1; ofs < 16; ofs <<= 1) m = fmaxf(m, __shfl_xor(m, ofs, 32));
    float sum = 0.f;
#pragma unroll
    for (int kt = 0; kt < 7; ++kt) {
      float e = __expf(s[kt][v] - m);
      s[kt][v] = e;
      sum += e;
    }
    for (int ofs = 1; ofs < 16; ofs <<= 1) sum += __shfl_xor(sum, ofs, 32);
    float inv = 1.0f / sum;
    bf16* pr = &P[wid * 2048 + (lanehi * 8 + v) * 128];
#pragma unroll
    for (int kt = 0; kt < 7; ++kt) pr[kt * 16 + lmod] = (bf16)(s[kt][v] * inv);
    pr[112 + lmod] = (bf16)0.f;          // pad keys -> zero prob
  }

  // ---- O = P @ V  (wave-local P; LDS ops are in-order within a wave) ----
  v8f oacc[4];
#pragma unroll
  for (int nt = 0; nt < 4; ++nt) oacc[nt] = {};
#pragma unroll
  for (int c = 0; c < 4; ++c) {
    const int c0 = c * 32;
    v16bf pf = make_frag(&P[wid * 2048 + lmod * 128 + c0], lanehi);
#pragma unroll
    for (int nt = 0; nt < 4; ++nt) {
      v16bf vf = make_frag(&Vt[(nt * 16 + lmod) * 128 + c0], lanehi);
      oacc[nt] = wmma_bf16(pf, vf, oacc[nt]);
    }
  }
#pragma unroll
  for (int nt = 0; nt < 4; ++nt)
#pragma unroll
    for (int v = 0; v < 8; ++v) {
      const int q = q0 + lanehi * 8 + v;
      if (q < 100)
        o[((size_t)(b * 100 + q)) * 768 + h * 64 + nt * 16 + lmod] =
            (bf16)oacc[nt][v];
    }
}

// ---------------------------------------------------------------------------
// Host-side orchestration
// ---------------------------------------------------------------------------
extern "C" void kernel_launch(void* const* d_in, const int* in_sizes, int n_in,
                              void* d_out, int out_size, void* d_ws,
                              size_t ws_size, hipStream_t stream) {
  (void)in_sizes; (void)n_in; (void)out_size; (void)ws_size;
  const float* nobs      = (const float*)d_in[0];
  const float* xwo       = (const float*)d_in[1];
  const float* obs_w     = (const float*)d_in[2];
  const float* obs_b     = (const float*)d_in[3];
  const float* word_w    = (const float*)d_in[4];
  const float* word_b    = (const float*)d_in[5];
  const float* pos_start = (const float*)d_in[6];
  const float* pos_1LC   = (const float*)d_in[7];
  const float* lvl_emb   = (const float*)d_in[8];
  const float* ada_w     = (const float*)d_in[9];
  const float* ada_b     = (const float*)d_in[10];
  const float* qkv_w     = (const float*)d_in[11];
  const float* qkv_b     = (const float*)d_in[12];
  const float* proj_w    = (const float*)d_in[13];
  const float* proj_b    = (const float*)d_in[14];
  const float* fc1_w     = (const float*)d_in[15];
  const float* fc1_b     = (const float*)d_in[16];
  const float* fc2_w     = (const float*)d_in[17];
  const float* fc2_b     = (const float*)d_in[18];
  const float* hnm_w     = (const float*)d_in[19];
  const float* hnm_b     = (const float*)d_in[20];
  const float* head_w    = (const float*)d_in[21];
  const float* head_b    = (const float*)d_in[22];

  const int B = 128, L = 100, C = 768, M = B * L;   // 12800 tokens

  // workspace partition (all 256B aligned)
  char* p = (char*)d_ws;
  auto take = [&](size_t bytes) {
    void* r = (void*)p;
    p += (bytes + 255) & ~(size_t)255;
    return r;
  };
  float* x      = (float*)take((size_t)M * C * 4);          // residual stream
  bf16*  hb1    = (bf16*) take((size_t)M * C * 2);          // ln out / attn out
  bf16*  hb2    = (bf16*) take((size_t)M * 4 * C * 2);      // gelu out
  bf16*  qkvb   = (bf16*) take((size_t)M * 3 * C * 2);
  float* cond   = (float*)take((size_t)B * C * 4);
  bf16*  conds  = (bf16*) take((size_t)B * C * 2);
  float* adabuf = (float*)take((size_t)B * 6 * C * 4);
  float* hnmbuf = (float*)take((size_t)B * 2 * C * 4);

  cond_kernel<<<B, 256, 0, stream>>>(nobs, obs_w, obs_b, cond, conds);
  embed_kernel<<<M, 256, 0, stream>>>(xwo, word_w, word_b, cond, pos_start,
                                      pos_1LC, lvl_emb, x);

  for (int i = 0; i < 8; ++i) {
    // AdaLN params: cond_s @ ada_w[i] + ada_b[i]  -> (B, 6C) f32
    gemm_wmma<3><<<dim3(6 * C / BN, B / BM), 256, 0, stream>>>(
        conds, ada_w + (size_t)i * C * 6 * C, ada_b + (size_t)i * 6 * C,
        adabuf, nullptr, nullptr, B, 6 * C, C, 6 * C, 0, 0);
    // h = ln(x)*(s1+1)+sh1
    ln_mod<<<M, 256, 0, stream>>>(x, hb1, adabuf, L, 6 * C, 2 * C, 4 * C);
    // qkv
    gemm_wmma<0><<<dim3(3 * C / BN, M / BM), 256, 0, stream>>>(
        hb1, qkv_w + (size_t)i * C * 3 * C, qkv_b + (size_t)i * 3 * C, qkvb,
        nullptr, nullptr, M, 3 * C, C, 3 * C, 0, 0);
    // attention (o -> hb1, bf16)
    attn_kernel<<<B * 12, 224, 0, stream>>>(qkvb, hb1);
    // x += (o @ proj + b) * g1
    gemm_wmma<2><<<dim3(C / BN, M / BM), 256, 0, stream>>>(
        hb1, proj_w + (size_t)i * C * C, proj_b + (size_t)i * C, nullptr, x,
        adabuf, M, C, C, C, 6 * C, 0);
    // h = ln(x)*(s2+1)+sh2
    ln_mod<<<M, 256, 0, stream>>>(x, hb1, adabuf, L, 6 * C, 3 * C, 5 * C);
    // gelu(h @ fc1 + b)
    gemm_wmma<1><<<dim3(4 * C / BN, M / BM), 256, 0, stream>>>(
        hb1, fc1_w + (size_t)i * C * 4 * C, fc1_b + (size_t)i * 4 * C, hb2,
        nullptr, nullptr, M, 4 * C, C, 4 * C, 0, 0);
    // x += (g @ fc2 + b) * g2
    gemm_wmma<2><<<dim3(C / BN, M / BM), 256, 0, stream>>>(
        hb2, fc2_w + (size_t)i * 4 * C * C, fc2_b + (size_t)i * C, nullptr, x,
        adabuf, M, C, 4 * C, C, 6 * C, C);
  }

  // head norm params + final LN + vocab projection
  gemm_wmma<3><<<dim3(2 * C / BN, B / BM), 256, 0, stream>>>(
      conds, hnm_w, hnm_b, hnmbuf, nullptr, nullptr, B, 2 * C, C, 2 * C, 0, 0);
  ln_mod<<<M, 256, 0, stream>>>(x, hb1, hnmbuf, L, 2 * C, 0, C);
  gemm_wmma<3><<<dim3(4096 / BN, M / BM), 256, 0, stream>>>(
      hb1, head_w, head_b, d_out, nullptr, nullptr, M, 4096, C, 4096, 0, 0);
}
